// NGPDensityField_80779744903209
// MI455X (gfx1250) — matrix operand order repaired and checked by reading, not compiled, not run on hardware
//
#include <hip/hip_runtime.h>
#include <hip/hip_bf16.h>
#include <math.h>

typedef __attribute__((ext_vector_type(16))) _Float16 v16h;
typedef __attribute__((ext_vector_type(8)))  float    v8f;

#define NLVL     5
#define LOG2_T   17
#define TSZ      (1 << LOG2_T)
#define ENC_DIM  10
#define HIDDEN   64

__device__ __forceinline__ unsigned pack2(float a, float b) {
    // single v_cvt_pk_rtz_f16_f32; builtin returns __fp16 ext_vector(2)
    auto p = __builtin_amdgcn_cvt_pkrtz(a, b);
    return __builtin_bit_cast(unsigned, p);
}

union Frag16 { unsigned u[8]; v16h h; };

__launch_bounds__(256)
__global__ void ngp_density_kernel(const float* __restrict__ positions,
                                   const float* __restrict__ aabb,
                                   const float* __restrict__ grid,
                                   const float* __restrict__ w1,
                                   const float* __restrict__ w2,
                                   float* __restrict__ out,
                                   int n)
{
    // resolutions: floor(16 * (128/16)^(l/4)) = {16,26,45,76,128}
    constexpr int  RES[NLVL]   = {16, 26, 45, 76, 128};
    constexpr bool DENSE[NLVL] = {true, true, true, false, false}; // (res+1)^3 <= T

    __shared__ unsigned enc_sh[8][32][6]; // packed f16 pairs, word5 = 0 pad
    __shared__ float    sel_sh[8][32];

    const int wv   = threadIdx.x >> 5;
    const int lane = threadIdx.x & 31;
    const int half = lane >> 4;     // 0: lanes 0-15, 1: lanes 16-31
    const int l16  = lane & 15;

    int gid = blockIdx.x * 256 + threadIdx.x;
    int gcl = gid < n ? gid : (n - 1);   // clamp loads; keep EXEC full for WMMA

    // ---------------- encoding: one point per lane ----------------
    const float mnx = aabb[0], mny = aabb[1], mnz = aabb[2];
    const float mxx = aabb[3], mxy = aabb[4], mxz = aabb[5];
    const float px = positions[3 * gcl + 0];
    const float py = positions[3 * gcl + 1];
    const float pz = positions[3 * gcl + 2];
    const float ux = (px - mnx) / (mxx - mnx);
    const float uy = (py - mny) / (mxy - mny);
    const float uz = (pz - mnz) / (mxz - mnz);
    const float sel = (ux > 0.f && ux < 1.f &&
                       uy > 0.f && uy < 1.f &&
                       uz > 0.f && uz < 1.f) ? 1.f : 0.f;

    float enc[ENC_DIM];
    #pragma unroll
    for (int l = 0; l < NLVL; ++l) {
        const float resf = (float)RES[l];
        const float sx = ux * resf, sy = uy * resf, sz = uz * resf;
        const float fx0 = floorf(sx), fy0 = floorf(sy), fz0 = floorf(sz);
        const float fx = sx - fx0, fy = sy - fy0, fz = sz - fz0;
        const unsigned x0 = (unsigned)(int)fx0;
        const unsigned y0 = (unsigned)(int)fy0;
        const unsigned z0 = (unsigned)(int)fz0;
        const float* tbl = grid + (size_t)l * (TSZ * 2);
        float a0 = 0.f, a1 = 0.f;
        #pragma unroll
        for (int c = 0; c < 8; ++c) {
            const unsigned cx = x0 + (c & 1);
            const unsigned cy = y0 + ((c >> 1) & 1);
            const unsigned cz = z0 + ((c >> 2) & 1);
            unsigned idx;
            if (DENSE[l]) {
                const unsigned r1 = (unsigned)(RES[l] + 1);
                idx = cx + cy * r1 + cz * r1 * r1;
                idx = idx < (unsigned)TSZ ? idx : (unsigned)(TSZ - 1); // JAX clamps OOB
            } else {
                idx = (cx * 1u) ^ (cy * 2654435761u) ^ (cz * 805459861u);
                idx &= (unsigned)(TSZ - 1);
            }
            const float wx = (c & 1)        ? fx : 1.f - fx;
            const float wy = ((c >> 1) & 1) ? fy : 1.f - fy;
            const float wz = ((c >> 2) & 1) ? fz : 1.f - fz;
            const float w = wx * wy * wz;
            const float f0 = tbl[2 * idx + 0];
            const float f1 = tbl[2 * idx + 1];
            a0 += w * f0;
            a1 += w * f1;
        }
        enc[2 * l + 0] = a0;
        enc[2 * l + 1] = a1;
    }

    // stash packed-f16 encoding + selector in LDS (wave-private slice)
    enc_sh[wv][lane][0] = pack2(enc[0], enc[1]);
    enc_sh[wv][lane][1] = pack2(enc[2], enc[3]);
    enc_sh[wv][lane][2] = pack2(enc[4], enc[5]);
    enc_sh[wv][lane][3] = pack2(enc[6], enc[7]);
    enc_sh[wv][lane][4] = pack2(enc[8], enc[9]);
    enc_sh[wv][lane][5] = 0u;
    sel_sh[wv][lane] = sel;
    asm volatile("s_wait_dscnt 0" ::: "memory"); // wave-local LDS RAW

    // ---------------- B fragments: w1^T tiles (32x16 f16, K=10 padded) ----------------
    // B layout: lanes 0-15 = column N (l16), v0..v3 = K0..7 pairs;
    //           lanes 16-31 v0..v3 = K8..15 (only K8,9 nonzero here).
    Frag16 B[4];
    #pragma unroll
    for (int t = 0; t < 4; ++t) {
        const float* r = w1 + (t * 16 + l16) * ENC_DIM;
        #pragma unroll
        for (int j = 0; j < 8; ++j) {
            unsigned v = 0u;
            if (j < 5) v = pack2(r[2 * j], r[2 * j + 1]);
            B[t].u[j] = half ? (j == 0 ? pack2(r[8], r[9]) : 0u) : (j < 4 ? v : 0u);
        }
    }
    float w2v[4];
    #pragma unroll
    for (int t = 0; t < 4; ++t) w2v[t] = w2[t * 16 + l16];

    // ---------------- two 16-point A tiles per wave ----------------
    #pragma unroll
    for (int at = 0; at < 2; ++at) {
        const int pt = at * 16 + l16;
        // A layout: lanes 0-15 row M=l16, v0..v3 = K0..7 pairs; lanes 16-31 v0..v3 = K8..15
        Frag16 A;
        A.u[0] = enc_sh[wv][pt][half ? 4 : 0];
        A.u[1] = enc_sh[wv][pt][half ? 5 : 1];
        A.u[2] = half ? 0u : enc_sh[wv][pt][2];
        A.u[3] = half ? 0u : enc_sh[wv][pt][3];
        A.u[4] = 0u; A.u[5] = 0u; A.u[6] = 0u; A.u[7] = 0u;

        float acc[8] = {0.f, 0.f, 0.f, 0.f, 0.f, 0.f, 0.f, 0.f};
        #pragma unroll
        for (int t = 0; t < 4; ++t) {
            v8f c = {0.f, 0.f, 0.f, 0.f, 0.f, 0.f, 0.f, 0.f};
            c = __builtin_amdgcn_wmma_f32_16x16x32_f16(
                    false, A.h, false, B[t].h, (short)0, c, false, false);
            // layer 2 fused: relu + scale by w2 (per-lane partial over N dim)
            #pragma unroll
            for (int j = 0; j < 8; ++j) {
                const float hj = c[j] > 0.f ? c[j] : 0.f;
                acc[j] += hj * w2v[t];
            }
        }
        // reduce over the 16 lanes of each half-wave (N dimension of C tile)
        #pragma unroll
        for (int m = 1; m < 16; m <<= 1) {
            #pragma unroll
            for (int j = 0; j < 8; ++j)
                acc[j] += __shfl_xor(acc[j], m, 32);
        }
        // C layout: vgpr j = row j (lanes 0-15) / row j+8 (lanes 16-31)
        if (l16 == 0) {
            const int rowbase = at * 16 + half * 8;
            const int base = blockIdx.x * 256 + wv * 32 + rowbase;
            #pragma unroll
            for (int j = 0; j < 8; ++j) {
                if (base + j < n) {
                    const float d = __expf(acc[j] - 1.f) * sel_sh[wv][rowbase + j];
                    out[base + j] = d;
                }
            }
        }
    }
}

extern "C" void kernel_launch(void* const* d_in, const int* in_sizes, int n_in,
                              void* d_out, int out_size, void* d_ws, size_t ws_size,
                              hipStream_t stream) {
    const float* positions = (const float*)d_in[0];
    const float* aabb      = (const float*)d_in[1];
    const float* grid      = (const float*)d_in[2];
    const float* w1        = (const float*)d_in[3];
    const float* w2        = (const float*)d_in[4];
    float* out = (float*)d_out;

    const int n = in_sizes[0] / 3;
    const int blocks = (n + 255) / 256;
    ngp_density_kernel<<<blocks, 256, 0, stream>>>(positions, aabb, grid, w1, w2, out, n);
}
